// EGCN_75960791598085
// MI455X (gfx1250) — compile-verified, matrix operand order checked by reading, not run on hardware
//
#include <hip/hip_runtime.h>
#include <hip/hip_bf16.h>

// ---------------- problem constants ----------------
#define T_STEPS 6
#define NN      50000
#define EE      800000
#define FINC    128   // fin for both layers; also the K dim of every GEMM
#define FOUT0   128
#define FOUT1   64

typedef __attribute__((ext_vector_type(16))) _Float16     v16h;
typedef __attribute__((ext_vector_type(8)))  float        v8f;
typedef __attribute__((ext_vector_type(4)))  unsigned int v4u;
typedef __attribute__((ext_vector_type(4)))  float        v4f;

// ---------------- WMMA fragment helper (16x16x32 f16, wave32) ----------------
// 16-bit operand layout (ISA 7.12.2): per lane, halves 0..7 hold K=kbase+g..+7
// and halves 8..15 hold K=kbase+16+g..+7 with g=(lane>>4)*8, for matrix
// row/col id = base + (lane&15).  With K-contiguous storage (ld = 128) a
// fragment is exactly two 16-byte contiguous chunks -> two b128 loads.
union FragU {
  struct { v4u lo, hi; } u;
  v16h h;
};

__device__ __forceinline__ v16h frag_ld(const _Float16* P, int rc_base,
                                        int kbase, int lane) {
  const int rc = rc_base + (lane & 15);
  const int g  = (lane >> 4) << 3;
  const _Float16* p = P + (size_t)rc * FINC + kbase + g;
  FragU f;
  f.u.lo = *(const v4u*)(p);
  f.u.hi = *(const v4u*)(p + 16);
  return f.h;
}

__device__ __forceinline__ v8f wmma16(v16h a, v16h b, v8f c) {
  return __builtin_amdgcn_wmma_f32_16x16x32_f16(false, a, false, b, (short)0, c,
                                                false, false);
}

// C/D 16x16 f32: VGPR r -> M = (lane>>4)*8 + r, N = lane&15.
__device__ __forceinline__ int cd_m(int lane, int r) { return ((lane >> 4) << 3) + r; }
__device__ __forceinline__ int cd_n(int lane) { return lane & 15; }

// ---------------- elementwise utility kernels ----------------
__global__ void cvt_f32_f16_kernel(const float* __restrict__ s,
                                   _Float16* __restrict__ d, int n) {
  int i = blockIdx.x * blockDim.x + threadIdx.x;
  if (i < n) d[i] = (_Float16)s[i];
}

// transpose + convert: S row-major [128 x cols] -> DT [cols x 128] f16
__global__ void cvtT_f32_f16_kernel(const float* __restrict__ s,
                                    _Float16* __restrict__ dt, int cols) {
  int i = blockIdx.x * blockDim.x + threadIdx.x;
  if (i >= FINC * cols) return;
  int r = i / cols;
  int c = i - r * cols;
  dt[(size_t)c * FINC + r] = (_Float16)s[i];
}

__global__ void copy_f32_kernel(const float* __restrict__ s,
                                float* __restrict__ d, int n) {
  int i = blockIdx.x * blockDim.x + threadIdx.x;
  if (i < n) d[i] = s[i];
}

__global__ void zero_f32_kernel(float* __restrict__ d, int n) {
  int i = blockIdx.x * blockDim.x + threadIdx.x;
  if (i < n) d[i] = 0.0f;
}

__global__ void relu_kernel(float* __restrict__ d, int n) {
  int i = blockIdx.x * blockDim.x + threadIdx.x;
  if (i < n) d[i] = fmaxf(d[i], 0.0f);
}

// RQhT[n][m] = rst[m][n] * Q[m][n]   (rst, Q row-major [128 x fout])
__global__ void rqT_kernel(const float* __restrict__ rst,
                           const float* __restrict__ Q,
                           _Float16* __restrict__ RQhT, int fout) {
  int i = blockIdx.x * blockDim.x + threadIdx.x;
  if (i >= FINC * fout) return;
  int m = i / fout;
  int n = i - m * fout;
  RQhT[(size_t)n * FINC + m] = (_Float16)(rst[i] * Q[i]);
}

// ---------------- scores = (X @ scorer)/||scorer|| + mask ----------------
__global__ void scores_kernel(const float* __restrict__ X,
                              const float* __restrict__ scorer,
                              const float* __restrict__ mask,
                              float* __restrict__ scores, int n) {
  __shared__ float sc[FINC];
  __shared__ float inv;
  const int tid = threadIdx.x;
  if (tid < FINC) sc[tid] = scorer[tid];
  __syncthreads();
  if (tid == 0) {
    float ss = 0.0f;
    for (int i = 0; i < FINC; ++i) ss += sc[i] * sc[i];
    inv = 1.0f / sqrtf(ss + 1e-30f);
  }
  __syncthreads();
  const int row = blockIdx.x * blockDim.x + tid;
  if (row < n) {
    const float* xr = X + (size_t)row * FINC;
    float d = 0.0f;
#pragma unroll 8
    for (int i = 0; i < FINC; ++i) d += xr[i] * sc[i];
    scores[row] = d * inv + mask[row];
  }
}

// ---------------- iterative top-k (single block, mutates scores) ----------------
__global__ void topk_kernel(float* __restrict__ scores, int* __restrict__ idx,
                            float* __restrict__ vals, int n, int k) {
  __shared__ float rv[1024];
  __shared__ int   ri[1024];
  const int tid = threadIdx.x;
  for (int j = 0; j < k; ++j) {
    float best = -3.0e38f;
    int   bi   = 0;
    for (int i = tid; i < n; i += 1024) {
      float s = scores[i];
      if (s > best) { best = s; bi = i; }
    }
    rv[tid] = best; ri[tid] = bi;
    __syncthreads();
    for (int off = 512; off > 0; off >>= 1) {
      if (tid < off && rv[tid + off] > rv[tid]) {
        rv[tid] = rv[tid + off]; ri[tid] = ri[tid + off];
      }
      __syncthreads();
    }
    if (tid == 0) {
      idx[j]  = ri[0];
      vals[j] = rv[0];
      scores[ri[0]] = -3.0e38f;   // exclude from later rounds
    }
    __syncthreads();
  }
}

// ---- zT[j][f] = X[idx[j]][f] * tanh(vals[j])  (K-contiguous for frag_ld) ----
__global__ void build_zT_kernel(const float* __restrict__ X,
                                const int* __restrict__ idx,
                                const float* __restrict__ vals,
                                _Float16* __restrict__ zhT, int k) {
  int t = blockIdx.x * blockDim.x + threadIdx.x;
  if (t >= k * FINC) return;
  int j = t >> 7;          // selected-node slot (N dim of the GEMM)
  int f = t & (FINC - 1);  // feature (K dim)
  float v = X[(size_t)idx[j] * FINC + f] * tanhf(vals[j]);
  zhT[t] = (_Float16)v;    // zhT[j*128 + f] : contiguous write
}

// ---------------- gates: upd = sigm(Wu@z + Uu@Q + bu), rst likewise ----------------
// M = K = 128, N = fout.  A matrices row-major [128x128] f16; B matrices stored
// transposed [N x 128] f16 (zhT, QhT).  One block, 8 waves, wave w owns M-tile w.
__global__ void __launch_bounds__(256)
gates_kernel(const _Float16* __restrict__ Wu, const _Float16* __restrict__ Uu,
             const _Float16* __restrict__ Wr, const _Float16* __restrict__ Ur,
             const _Float16* __restrict__ zhT, const _Float16* __restrict__ QhT,
             const float* __restrict__ bu, const float* __restrict__ br,
             float* __restrict__ upd, float* __restrict__ rst, int fout) {
  const int lane  = threadIdx.x & 31;
  const int wave  = threadIdx.x >> 5;
  const int mbase = wave << 4;
  const int ntiles = fout >> 4;
  for (int nt = 0; nt < ntiles; ++nt) {
    const int nbase = nt << 4;
    v8f au = {};
    v8f ar = {};
#pragma unroll
    for (int kt = 0; kt < 4; ++kt) {
      const int kb = kt << 5;
      v16h bz = frag_ld(zhT, nbase, kb, lane);
      v16h bq = frag_ld(QhT, nbase, kb, lane);
      v16h a;
      a = frag_ld(Wu, mbase, kb, lane); au = wmma16(a, bz, au);
      a = frag_ld(Uu, mbase, kb, lane); au = wmma16(a, bq, au);
      a = frag_ld(Wr, mbase, kb, lane); ar = wmma16(a, bz, ar);
      a = frag_ld(Ur, mbase, kb, lane); ar = wmma16(a, bq, ar);
    }
#pragma unroll
    for (int r = 0; r < 8; ++r) {
      const int m = mbase + cd_m(lane, r);
      const size_t o = (size_t)m * fout + nbase + cd_n(lane);
      float u  = au[r] + bu[o];
      float rr = ar[r] + br[o];
      upd[o] = 1.0f / (1.0f + __expf(-u));
      rst[o] = 1.0f / (1.0f + __expf(-rr));
    }
  }
}

// ---------------- h_cap + GRU blend: Qn = (1-u)*Q + u*tanh(Wh@z + Uh@RQ + bh) ----
// Emits Qn (f32 row-major, recurrence) and QnhT (f16 [fout x 128], for X@Qn).
__global__ void __launch_bounds__(256)
hcap_kernel(const _Float16* __restrict__ Wh, const _Float16* __restrict__ Uh,
            const _Float16* __restrict__ zhT, const _Float16* __restrict__ RQhT,
            const float* __restrict__ bh, const float* __restrict__ upd,
            const float* __restrict__ Q, float* __restrict__ Qn,
            _Float16* __restrict__ QnhT, int fout) {
  const int lane  = threadIdx.x & 31;
  const int wave  = threadIdx.x >> 5;
  const int mbase = wave << 4;
  const int ntiles = fout >> 4;
  for (int nt = 0; nt < ntiles; ++nt) {
    const int nbase = nt << 4;
    v8f ah = {};
#pragma unroll
    for (int kt = 0; kt < 4; ++kt) {
      const int kb = kt << 5;
      v16h a, b;
      a = frag_ld(Wh, mbase, kb, lane);
      b = frag_ld(zhT, nbase, kb, lane);
      ah = wmma16(a, b, ah);
      a = frag_ld(Uh, mbase, kb, lane);
      b = frag_ld(RQhT, nbase, kb, lane);
      ah = wmma16(a, b, ah);
    }
#pragma unroll
    for (int r = 0; r < 8; ++r) {
      const int m  = mbase + cd_m(lane, r);
      const int nn = nbase + cd_n(lane);
      const size_t o = (size_t)m * fout + nn;
      float h = tanhf(ah[r] + bh[o]);
      float u = upd[o];
      float q = Q[o];
      float qn = (1.0f - u) * q + u * h;
      Qn[o] = qn;
      QnhT[(size_t)nn * FINC + m] = (_Float16)qn;
    }
  }
}

// ---------------- XW = X @ Qn : [N,128] x [128,fout], f16 WMMA via LDS ----------
__global__ void __launch_bounds__(256)
xqn_kernel(const float* __restrict__ X, const _Float16* __restrict__ QnhT,
           float* __restrict__ XW, int n, int fout) {
  __shared__ __align__(16) _Float16 xt[128 * FINC];   // 32 KB f16 tile of X
  const int tid   = threadIdx.x;
  const int rbase = blockIdx.x << 7;
  // vectorized f32->f16 tile staging: 4096 float4 reads, 256 thr x 16 iters
  const v4f* X4 = (const v4f*)(X);
  for (int i = tid; i < 128 * FINC / 4; i += 256) {
    const int r   = i >> 5;        // /32 float4s per row
    const int c4  = i & 31;
    const int row = rbase + r;
    v4f v = {0.0f, 0.0f, 0.0f, 0.0f};
    if (row < n) v = X4[(size_t)row * (FINC / 4) + c4];
    _Float16* dst = xt + r * FINC + c4 * 4;
    dst[0] = (_Float16)v.x; dst[1] = (_Float16)v.y;
    dst[2] = (_Float16)v.z; dst[3] = (_Float16)v.w;
  }
  __syncthreads();
  const int lane  = tid & 31;
  const int wave  = tid >> 5;
  const int mbase = wave << 4;
  const int ntiles = fout >> 4;
  for (int nt = 0; nt < ntiles; ++nt) {
    const int nbase = nt << 4;
    v8f acc = {};
#pragma unroll
    for (int kt = 0; kt < 4; ++kt) {
      v16h a = frag_ld(xt, mbase, kt << 5, lane);      // ds_load_b128 x2
      v16h b = frag_ld(QnhT, nbase, kt << 5, lane);    // global b128 x2 (L2 hot)
      acc = wmma16(a, b, acc);
    }
#pragma unroll
    for (int r = 0; r < 8; ++r) {
      const int row = rbase + mbase + cd_m(lane, r);
      if (row < n) XW[(size_t)row * fout + nbase + cd_n(lane)] = acc[r];
    }
  }
}

// ---------------- edge message + segment sum: Y[r] += v * XW[c] ------------------
// one wave32 per edge; lanes stride over fout channels (coalesced).
__global__ void __launch_bounds__(256)
edge_kernel(const int* __restrict__ rows, const int* __restrict__ cols,
            const float* __restrict__ vals, const float* __restrict__ XW,
            float* __restrict__ Y, int e_count, int fout) {
  const int lane = threadIdx.x & 31;
  const int e = blockIdx.x * (blockDim.x >> 5) + (threadIdx.x >> 5);
  if (e >= e_count) return;
  const int   r = rows[e];
  const int   c = cols[e];
  const float v = vals[e];
  const float* src = XW + (size_t)c * fout;
  float*       dst = Y + (size_t)r * fout;
  __builtin_prefetch(src + lane, 0, 0);   // global_prefetch_b8: hide gather latency
  for (int col = lane; col < fout; col += 32)
    atomicAdd(&dst[col], v * src[col]);
}

// ======================= host orchestration =======================
struct LayerPtrs {
  const float *scorer, *Wu, *Uu, *bu, *Wr, *Ur, *br, *Wh, *Uh, *bh, *W_init;
};

static inline int cdiv(int a, int b) { return (a + b - 1) / b; }

extern "C" void kernel_launch(void* const* d_in, const int* in_sizes, int n_in,
                              void* d_out, int out_size, void* d_ws, size_t ws_size,
                              hipStream_t stream) {
  (void)in_sizes; (void)n_in; (void)out_size; (void)ws_size;

  const int*   A_rows = (const int*)d_in[0];     // [T,E]
  const int*   A_cols = (const int*)d_in[1];     // [T,E]
  const float* A_vals = (const float*)d_in[2];   // [T,E]
  const float* Nodes  = (const float*)d_in[3];   // [T,N,128]
  const float* masks  = (const float*)d_in[4];   // [T,N,1]

  // params flattened in setup_inputs() dict order: layer0 then layer1, each
  // {scorer, Wu, Uu, bu, Wr, Ur, br, Wh, Uh, bh, W_init}
  LayerPtrs L[2];
  for (int l = 0; l < 2; ++l) {
    int b = 5 + l * 11;
    L[l].scorer = (const float*)d_in[b + 0];
    L[l].Wu = (const float*)d_in[b + 1];
    L[l].Uu = (const float*)d_in[b + 2];
    L[l].bu = (const float*)d_in[b + 3];
    L[l].Wr = (const float*)d_in[b + 4];
    L[l].Ur = (const float*)d_in[b + 5];
    L[l].br = (const float*)d_in[b + 6];
    L[l].Wh = (const float*)d_in[b + 7];
    L[l].Uh = (const float*)d_in[b + 8];
    L[l].bh = (const float*)d_in[b + 9];
    L[l].W_init = (const float*)d_in[b + 10];
  }

  // ---- carve workspace ----
  char* w = (char*)d_ws;
  auto carve = [&](size_t bytes) {
    void* p = (void*)w;
    w += (bytes + 255) & ~(size_t)255;
    return p;
  };
  float* Y0     = (float*)carve((size_t)NN * 128 * 4);   // layer0 output at t
  float* XW     = (float*)carve((size_t)NN * 128 * 4);   // X @ Qn scratch
  float* scores = (float*)carve((size_t)NN * 4);
  int*   idx    = (int*)carve(128 * 4);
  float* vals   = (float*)carve(128 * 4);
  float* upd    = (float*)carve(128 * 128 * 4);
  float* rstb   = (float*)carve(128 * 128 * 4);
  float* Q0[2]  = {(float*)carve(128 * 128 * 4), (float*)carve(128 * 128 * 4)};
  float* Q1[2]  = {(float*)carve(128 * 64 * 4), (float*)carve(128 * 64 * 4)};
  _Float16* zhT  = (_Float16*)carve(128 * 128 * 2);  // [k x 128]
  _Float16* QhT  = (_Float16*)carve(128 * 128 * 2);  // [fout x 128]
  _Float16* QnhT = (_Float16*)carve(128 * 128 * 2);  // [fout x 128]
  _Float16* RQhT = (_Float16*)carve(128 * 128 * 2);  // [fout x 128]
  // f16 copies of the 6 recurrent matrices per layer (row-major, ld=128)
  _Float16* Wh16[2][6];
  for (int l = 0; l < 2; ++l)
    for (int m = 0; m < 6; ++m) Wh16[l][m] = (_Float16*)carve(128 * 128 * 2);

  const int NM = 128 * 128;  // 16384
  const dim3 B256(256);

  // ---- convert weights to f16 (every call: deterministic, no static state) ----
  for (int l = 0; l < 2; ++l) {
    const float* src[6] = {L[l].Wu, L[l].Uu, L[l].Wr, L[l].Ur, L[l].Wh, L[l].Uh};
    for (int m = 0; m < 6; ++m)
      cvt_f32_f16_kernel<<<cdiv(NM, 256), B256, 0, stream>>>(src[m], Wh16[l][m], NM);
  }

  // ---- init Q = W_init ----
  copy_f32_kernel<<<cdiv(128 * 128, 256), B256, 0, stream>>>(L[0].W_init, Q0[0], 128 * 128);
  copy_f32_kernel<<<cdiv(128 * 64, 256), B256, 0, stream>>>(L[1].W_init, Q1[0], 128 * 64);

  float* out = (float*)d_out;  // [N, 64]

  for (int t = 0; t < T_STEPS; ++t) {
    const float* Xt = Nodes + (size_t)t * NN * FINC;
    const float* mt = masks + (size_t)t * NN;
    const int* er = A_rows + (size_t)t * EE;
    const int* ec = A_cols + (size_t)t * EE;
    const float* ev = A_vals + (size_t)t * EE;

    // ================= layer 0 (fin=128, fout=128) =================
    {
      const int fout = FOUT0, nq = 128 * fout;
      float* Qcur = Q0[t & 1];
      float* Qnxt = Q0[(t + 1) & 1];
      scores_kernel<<<cdiv(NN, 256), B256, 0, stream>>>(Xt, L[0].scorer, mt, scores, NN);
      topk_kernel<<<1, 1024, 0, stream>>>(scores, idx, vals, NN, fout);
      build_zT_kernel<<<cdiv(fout * FINC, 256), B256, 0, stream>>>(Xt, idx, vals, zhT, fout);
      cvtT_f32_f16_kernel<<<cdiv(nq, 256), B256, 0, stream>>>(Qcur, QhT, fout);
      gates_kernel<<<1, B256, 0, stream>>>(Wh16[0][0], Wh16[0][1], Wh16[0][2], Wh16[0][3],
                                           zhT, QhT, L[0].bu, L[0].br, upd, rstb, fout);
      rqT_kernel<<<cdiv(nq, 256), B256, 0, stream>>>(rstb, Qcur, RQhT, fout);
      hcap_kernel<<<1, B256, 0, stream>>>(Wh16[0][4], Wh16[0][5], zhT, RQhT, L[0].bh,
                                          upd, Qcur, Qnxt, QnhT, fout);
      xqn_kernel<<<cdiv(NN, 128), B256, 0, stream>>>(Xt, QnhT, XW, NN, fout);
      zero_f32_kernel<<<cdiv(NN * fout, 256), B256, 0, stream>>>(Y0, NN * fout);
      edge_kernel<<<cdiv(EE, 8), B256, 0, stream>>>(er, ec, ev, XW, Y0, EE, fout);
      relu_kernel<<<cdiv(NN * fout, 256), B256, 0, stream>>>(Y0, NN * fout);
    }

    // ================= layer 1 (fin=128, fout=64) =================
    {
      const int fout = FOUT1, nq = 128 * fout;
      float* Qcur = Q1[t & 1];
      float* Qnxt = Q1[(t + 1) & 1];
      scores_kernel<<<cdiv(NN, 256), B256, 0, stream>>>(Y0, L[1].scorer, mt, scores, NN);
      topk_kernel<<<1, 1024, 0, stream>>>(scores, idx, vals, NN, fout);
      build_zT_kernel<<<cdiv(fout * FINC, 256), B256, 0, stream>>>(Y0, idx, vals, zhT, fout);
      cvtT_f32_f16_kernel<<<cdiv(nq, 256), B256, 0, stream>>>(Qcur, QhT, fout);
      gates_kernel<<<1, B256, 0, stream>>>(Wh16[1][0], Wh16[1][1], Wh16[1][2], Wh16[1][3],
                                           zhT, QhT, L[1].bu, L[1].br, upd, rstb, fout);
      rqT_kernel<<<cdiv(nq, 256), B256, 0, stream>>>(rstb, Qcur, RQhT, fout);
      hcap_kernel<<<1, B256, 0, stream>>>(Wh16[1][4], Wh16[1][5], zhT, RQhT, L[1].bh,
                                          upd, Qcur, Qnxt, QnhT, fout);
      if (t == T_STEPS - 1) {
        // only the final timestep's output is returned (h[-1])
        xqn_kernel<<<cdiv(NN, 128), B256, 0, stream>>>(Y0, QnhT, XW, NN, fout);
        zero_f32_kernel<<<cdiv(NN * fout, 256), B256, 0, stream>>>(out, NN * fout);
        edge_kernel<<<cdiv(EE, 8), B256, 0, stream>>>(er, ec, ev, XW, out, EE, fout);
        relu_kernel<<<cdiv(NN * fout, 256), B256, 0, stream>>>(out, NN * fout);
      }
    }
  }
}
